// VRWKV_SpatialMix_Tri_Eff_2D_6923487281167
// MI455X (gfx1250) — matrix operand order repaired and checked by reading, not compiled, not run on hardware
//
#include <hip/hip_runtime.h>
#include <hip/hip_bf16.h>
#include <cstdint>

typedef __attribute__((ext_vector_type(16))) __bf16 v16bf;
typedef __attribute__((ext_vector_type(8)))  float  v8f;

namespace {
constexpr int Bb = 16, Hh = 96, Ww = 96, Cc = 192;
constexpr int Tt = Hh * Ww;             // 9216
constexpr int BT = Bb * Tt;             // 147456 rows in the GEMMs
constexpr int NWAVE = 8;                // waves per block (256 threads, wave32)
constexpr int NT = 4;                   // 16x16 n-tiles per wave -> 64 cols
constexpr int LCH = 128;                // wkv chunk length (divides Tt)
constexpr int NCHUNK = Tt / LCH;        // 72 chunks per sequence
constexpr int NSTATE = Bb * Cc * NCHUNK;
constexpr int WELEM = Cc * Cc;          // elements per weight matrix
}

// Pack 16 contiguous floats (already in registers) to v16bf.
__device__ inline void cvt16(const float* ff, v16bf& dst) {
#pragma unroll
  for (int i = 0; i < 16; ++i) dst[i] = (__bf16)ff[i];
}

// ---------------------------------------------------------------------------
// One-time weight pre-pack: f32 (C,C) row-major -> bf16, concatenated
// [Wk|Wv|Wr|Wo]. Lets the GEMM hot loops load v16bf weight tiles directly
// (2x b128, zero converts) instead of 4x b128 + 8x v_cvt_pk_bf16_f32.
// ---------------------------------------------------------------------------
__global__ void pack_weights_bf16(const float* __restrict__ Wk,
                                  const float* __restrict__ Wv,
                                  const float* __restrict__ Wr,
                                  const float* __restrict__ Wo,
                                  __bf16* __restrict__ out)
{
  const int idx = blockIdx.x * blockDim.x + threadIdx.x;
  if (idx >= 4 * WELEM) return;
  const int m = idx / WELEM, e = idx % WELEM;
  const float* src = (m == 0) ? Wk : (m == 1) ? Wv : (m == 2) ? Wr : Wo;
  out[idx] = (__bf16)src[e];
}

// ---------------------------------------------------------------------------
// Fused projection GEMM: {k, v, sigmoid(r)} = shift2d(x) @ {Wk,Wv,Wr}^T.
// bf16 WMMA, fp32 accumulate. Each A tile (16x32) is loaded/converted ONCE
// and feeds 12 WMMAs (3 matrices x 4 n-tiles) -> x is read once total.
// Weights are pre-packed bf16; row bases hoisted, K offsets are immediates.
// ---------------------------------------------------------------------------
__global__ void __launch_bounds__(256)
gemm3_shift_bf16(const float* __restrict__ X,
                 const __bf16* __restrict__ Wp,     // [Wk|Wv|Wr] bf16
                 float* __restrict__ Yk, float* __restrict__ Yv,
                 float* __restrict__ Yr)
{
  const int lane  = threadIdx.x & 31;
  const int wave  = threadIdx.x >> 5;
  const int strip = blockIdx.x * NWAVE + wave;   // M strip
  const int m0    = strip * 16;
  const int n0    = blockIdx.y * (NT * 16);
  const int mrow  = m0 + (lane & 15);            // this lane's token row
  const int khalf = lane >> 4;                   // 0: K lo-half, 1: K hi-half
  const int jl    = lane & 15;

  const int tb = mrow % Tt;                      // token within its batch
  const int hh = tb / Ww;
  const int ww = tb % Ww;

  // Precompute the 4 directional source rows for the quarter-channel shift.
  // (never dereferenced when the valid flag is false)
  const float* arow[4];
  bool aok[4];
  arow[0] = X + (size_t)(mrow - 1)  * Cc; aok[0] = (ww > 0);       // from w-1
  arow[1] = X + (size_t)(mrow + 1)  * Cc; aok[1] = (ww < Ww - 1);  // from w+1
  arow[2] = X + (size_t)(mrow - Ww) * Cc; aok[2] = (hh > 0);       // from h-1
  arow[3] = X + (size_t)(mrow + Ww) * Cc; aok[3] = (hh < Hh - 1);  // from h+1

  // Hoisted bf16 weight-row base pointers (K-invariant).
  const __bf16* bbase[3][NT];
#pragma unroll
  for (int nt = 0; nt < NT; ++nt) {
    const size_t jr = (size_t)(n0 + nt * 16 + jl) * Cc + khalf * 16;
#pragma unroll
    for (int m = 0; m < 3; ++m) bbase[m][nt] = Wp + (size_t)m * WELEM + jr;
  }

  v8f acc[3][NT];
#pragma unroll
  for (int m = 0; m < 3; ++m)
#pragma unroll
    for (int i = 0; i < NT; ++i) acc[m][i] = v8f{0.f,0.f,0.f,0.f,0.f,0.f,0.f,0.f};

#pragma unroll
  for (int kb = 0; kb < Cc; kb += 32) {
    // ---- A tile (16x32 bf16): lane holds row mrow, two 8-wide K runs.
    // Layout: lanes 0-15 K {0..15}, lanes 16-31 K {16..31} (matches B).
    float fa[16];
#pragma unroll
    for (int r = 0; r < 2; ++r) {
      const int c0  = kb + khalf * 16 + r * 8;   // depends on khalf (lane half)
      const int grp = c0 / (Cc / 4);             // shift group, depends on khalf
      float4 f[2];
      if (aok[grp]) {
        const float4* p = reinterpret_cast<const float4*>(arow[grp] + c0);
        f[0] = p[0]; f[1] = p[1];                // 2x global_load_b128
      } else {
        f[0] = make_float4(0.f,0.f,0.f,0.f); f[1] = f[0];
      }
      const float* ff = reinterpret_cast<const float*>(f);
#pragma unroll
      for (int i = 0; i < 8; ++i) fa[r * 8 + i] = ff[i];
    }
    v16bf a;
    cvt16(fa, a);

    // ---- 3 matrices x NT tiles of bf16 B (32x16) + WMMA, reusing A.
#pragma unroll
    for (int m = 0; m < 3; ++m) {
#pragma unroll
      for (int nt = 0; nt < NT; ++nt) {
        const v16bf b = *reinterpret_cast<const v16bf*>(bbase[m][nt] + kb);
        acc[m][nt] = __builtin_amdgcn_wmma_f32_16x16x32_bf16(
            false, a, false, b, (short)0, acc[m][nt], false, false);
      }
    }
  }

  // ---- store: D VGPR r -> row m0 + r + 8*khalf, col = n0 + nt*16 + lane%16
#pragma unroll
  for (int nt = 0; nt < NT; ++nt) {
#pragma unroll
    for (int r = 0; r < 8; ++r) {
      const size_t row = (size_t)(m0 + r + khalf * 8);
      const size_t at  = row * Cc + n0 + nt * 16 + jl;
      Yk[at] = acc[0][nt][r];
      Yv[at] = acc[1][nt][r];
      Yr[at] = 1.f / (1.f + __expf(-acc[2][nt][r]));   // fused sigmoid
    }
  }
}

// ---------------------------------------------------------------------------
// Single-matrix GEMM for the output projection (no shift), bf16 weights.
// ---------------------------------------------------------------------------
__global__ void __launch_bounds__(256)
gemm_bf16(const float* __restrict__ X, const __bf16* __restrict__ Wp,
          float* __restrict__ Y)
{
  const int lane  = threadIdx.x & 31;
  const int wave  = threadIdx.x >> 5;
  const int strip = blockIdx.x * NWAVE + wave;
  const int m0    = strip * 16;
  const int n0    = blockIdx.y * (NT * 16);
  const int mrow  = m0 + (lane & 15);
  const int khalf = lane >> 4;
  const int jl    = lane & 15;

  const float* xrow = X + (size_t)mrow * Cc;
  const __bf16* bbase[NT];
#pragma unroll
  for (int nt = 0; nt < NT; ++nt)
    bbase[nt] = Wp + (size_t)(n0 + nt * 16 + jl) * Cc + khalf * 16;

  v8f acc[NT];
#pragma unroll
  for (int i = 0; i < NT; ++i) acc[i] = v8f{0.f,0.f,0.f,0.f,0.f,0.f,0.f,0.f};

#pragma unroll
  for (int kb = 0; kb < Cc; kb += 32) {
    float fa[16];
#pragma unroll
    for (int r = 0; r < 2; ++r) {
      const int c0 = kb + khalf * 16 + r * 8;
      float4 f[2];
      const float4* p = reinterpret_cast<const float4*>(xrow + c0);
      f[0] = p[0]; f[1] = p[1];
      const float* ff = reinterpret_cast<const float*>(f);
#pragma unroll
      for (int i = 0; i < 8; ++i) fa[r * 8 + i] = ff[i];
    }
    v16bf a;
    cvt16(fa, a);

#pragma unroll
    for (int nt = 0; nt < NT; ++nt) {
      const v16bf b = *reinterpret_cast<const v16bf*>(bbase[nt] + kb);
      acc[nt] = __builtin_amdgcn_wmma_f32_16x16x32_bf16(
          false, a, false, b, (short)0, acc[nt], false, false);
    }
  }

#pragma unroll
  for (int nt = 0; nt < NT; ++nt) {
#pragma unroll
    for (int r = 0; r < 8; ++r) {
      const size_t row = (size_t)(m0 + r + khalf * 8);
      Y[row * Cc + n0 + nt * 16 + jl] = acc[nt][r];
    }
  }
}

// ---------------------------------------------------------------------------
// WKV chunked parallel scan. State (p,q,o): numerator, denominator, log-max.
// ---------------------------------------------------------------------------
__global__ void wkv_pass1(const float* __restrict__ kq, const float* __restrict__ vb,
                          const float* __restrict__ sd,
                          float* __restrict__ ps, float* __restrict__ qs,
                          float* __restrict__ os)
{
  const int idx = blockIdx.x * blockDim.x + threadIdx.x;   // g*Cc + c
  if (idx >= NSTATE) return;
  const int c  = idx % Cc;
  const int g  = idx / Cc;               // global chunk (LCH | Tt -> 1 batch)
  const int t0 = g * LCH;
  const float w = sd[c] * (1.0f / Tt);
  float p = 0.f, q = 0.f, o = -1e38f;
  const float* kp = kq + (size_t)t0 * Cc + c;
  const float* vp = vb + (size_t)t0 * Cc + c;
  for (int t = 0; t < LCH; ++t) {
    const float kt = kp[(size_t)t * Cc];
    const float vt = vp[(size_t)t * Cc];
    const float no2 = fmaxf(w + o, kt);
    const float A2  = __expf(w + o - no2);
    const float B2  = __expf(kt - no2);
    p = A2 * p + B2 * vt;
    q = A2 * q + B2;
    o = no2;
  }
  ps[idx] = p; qs[idx] = q; os[idx] = o;
}

__global__ void wkv_pass2(const float* __restrict__ sd,
                          float* __restrict__ ps, float* __restrict__ qs,
                          float* __restrict__ os)
{
  const int idx = blockIdx.x * blockDim.x + threadIdx.x;   // b*Cc + c
  if (idx >= Bb * Cc) return;
  const int c = idx % Cc, b = idx / Cc;
  const float w  = sd[c] * (1.0f / Tt);
  const float Lw = w * (float)LCH;       // L decay steps in log space
  float p = 0.f, q = 0.f, o = -1e38f;    // exclusive prefix before chunk 0
  for (int ch = 0; ch < NCHUNK; ++ch) {
    const size_t si = ((size_t)(b * NCHUNK + ch)) * Cc + c;
    const float pl = ps[si], ql = qs[si], ol = os[si];
    ps[si] = p; qs[si] = q; os[si] = o;  // write exclusive prefix state
    const float o1 = o + Lw;             // decay prefix across this chunk
    const float no = fmaxf(o1, ol);
    const float e1 = __expf(o1 - no);
    const float e2 = __expf(ol - no);
    p = e1 * p + e2 * pl;
    q = e1 * q + e2 * ql;
    o = no;
  }
}

// pass3 replays each chunk from its exclusive prefix, emitting sy = sr*y.
// sy aliases the k buffer (read-before-write within each thread) -> no
// __restrict__ on those pointers.
__global__ void wkv_pass3(const float* kq, const float* __restrict__ vb,
                          const float* __restrict__ srb,
                          const float* __restrict__ sd, const float* __restrict__ sf,
                          const float* __restrict__ ps, const float* __restrict__ qs,
                          const float* __restrict__ os, float* sy)
{
  const int idx = blockIdx.x * blockDim.x + threadIdx.x;
  if (idx >= NSTATE) return;
  const int c  = idx % Cc;
  const int g  = idx / Cc;
  const int t0 = g * LCH;
  const float w = sd[c] * (1.0f / Tt);
  const float u = sf[c] * (1.0f / Tt);
  float p = ps[idx], q = qs[idx], o = os[idx];
  const size_t base = (size_t)t0 * Cc + c;
  for (int t = 0; t < LCH; ++t) {
    const size_t at = base + (size_t)t * Cc;
    const float kt = kq[at];
    const float vt = vb[at];
    const float st = srb[at];
    // output from state *before* this token
    const float no = fmaxf(o, u + kt);
    const float A  = __expf(o - no);
    const float Bc = __expf(u + kt - no);
    const float y  = (A * p + Bc * vt) / (A * q + Bc);
    sy[at] = st * y;
    // state update
    const float no2 = fmaxf(w + o, kt);
    const float A2  = __expf(w + o - no2);
    const float B2  = __expf(kt - no2);
    p = A2 * p + B2 * vt;
    q = A2 * q + B2;
    o = no2;
  }
}

// ---------------------------------------------------------------------------
extern "C" void kernel_launch(void* const* d_in, const int* in_sizes, int n_in,
                              void* d_out, int out_size, void* d_ws, size_t ws_size,
                              hipStream_t stream) {
  const float* x  = (const float*)d_in[0];   // (B,T,C)
  const float* Wk = (const float*)d_in[1];   // (C,C)
  const float* Wv = (const float*)d_in[2];
  const float* Wr = (const float*)d_in[3];
  const float* Wo = (const float*)d_in[4];
  const float* sd = (const float*)d_in[5];   // (3,C), row 0 used
  const float* sf = (const float*)d_in[6];   // (3,C), row 0 used
  float* out = (float*)d_out;

  float* ws    = (float*)d_ws;
  float* kbuf  = ws;                               // BT*C f32
  float* vbuf  = kbuf  + (size_t)BT * Cc;          // BT*C f32
  float* srbuf = vbuf  + (size_t)BT * Cc;          // BT*C f32
  float* ps    = srbuf + (size_t)BT * Cc;          // NSTATE f32
  float* qs    = ps + NSTATE;
  float* os    = qs + NSTATE;
  __bf16* wpack = (__bf16*)(os + NSTATE);          // 4*C*C bf16 [Wk|Wv|Wr|Wo]
  (void)in_sizes; (void)n_in; (void)out_size; (void)ws_size;

  const dim3 gblk(256);
  const dim3 ggrid(BT / (16 * NWAVE), Cc / (NT * 16));   // (1152, 3)

  // one-time bf16 weight pre-pack (0.3 MB, L2-resident thereafter)
  pack_weights_bf16<<<(4 * WELEM + 255) / 256, 256, 0, stream>>>(Wk, Wv, Wr, Wo, wpack);

  // k, v, sigmoid(r) in ONE pass over x (fused shift, shared A tiles)
  gemm3_shift_bf16<<<ggrid, gblk, 0, stream>>>(x, wpack, kbuf, vbuf, srbuf);

  // WKV: chunked associative scan (72x parallelism per sequence-channel)
  wkv_pass1<<<(NSTATE + 255) / 256, 256, 0, stream>>>(kbuf, vbuf, sd, ps, qs, os);
  wkv_pass2<<<(Bb * Cc + 255) / 256, 256, 0, stream>>>(sd, ps, qs, os);
  wkv_pass3<<<(NSTATE + 255) / 256, 256, 0, stream>>>(kbuf, vbuf, srbuf, sd, sf,
                                                      ps, qs, os, /*sy=*/kbuf);

  // out = (sr*y) @ Wo^T
  gemm_bf16<<<ggrid, gblk, 0, stream>>>(kbuf, wpack + (size_t)3 * WELEM, out);
}
